// SVDQuantLinear_87849261072831
// MI455X (gfx1250) — matrix operand order repaired and checked by reading, not compiled, not run on hardware
//
#include <hip/hip_runtime.h>

typedef _Float16 v16h __attribute__((ext_vector_type(16)));
typedef _Float16 v8h  __attribute__((ext_vector_type(8)));
typedef _Float16 v2h  __attribute__((ext_vector_type(2)));
typedef float    v8f  __attribute__((ext_vector_type(8)));
typedef int      v4i  __attribute__((ext_vector_type(4)));
typedef unsigned int u32;
typedef u32      v8u  __attribute__((ext_vector_type(8)));

#define IN_F   8192
#define OUT_F  28672
#define RANK   32
#define GROUP  64
#define MTOK   16
#define SCALES_PER_ROW (IN_F / GROUP)   // 128

// ---------------------------------------------------------------------------
// Kernel 1: t[m][r] = sum_k x[m,k] * lora_down[r,k]  -> 16x32 fp16 in d_ws.
// ---------------------------------------------------------------------------
__global__ __launch_bounds__(512) void lora_down_kernel(
    const _Float16* __restrict__ x,
    const _Float16* __restrict__ ld,
    _Float16* __restrict__ t)
{
  int tid = threadIdx.x;
  int m = tid >> 5;        // 0..15 token
  int r = tid & 31;        // 0..31 rank
  const _Float16* xr = x  + (size_t)m * IN_F;
  const _Float16* lr = ld + (size_t)r * IN_F;
  float acc = 0.0f;
  for (int k = 0; k < IN_F; k += 8) {
    v8h a = *(const v8h*)(xr + k);
    v8h b = *(const v8h*)(lr + k);
#pragma unroll
    for (int j = 0; j < 8; ++j) acc += (float)a[j] * (float)b[j];
  }
  t[m * RANK + r] = (_Float16)acc;
}

// ---------------------------------------------------------------------------
// Kernel 2: one wave per 16x16 output tile.
//   acc  = WMMA(t, lora_up tile)          (K = RANK = 32 seed)
//   acc += sum_K WMMA(x tile, dequant(qweight) tile)
//   out  = acc + bias
//
// Dequant trick (exact): q in [-8,7]  =>  (q & 0xF) ^ 8 == q + 8, and
// f16 bits (0x6408 ^ (q & 0xF)) == f16(1032 + q) since ulp(1024) = 1.
// Then (h - 1032) is exact in f16 and *scale matches the reference's single
// f16 rounding of q*s. All ops are packed 2-wide 32-bit VALU.
// ---------------------------------------------------------------------------
__global__ __launch_bounds__(256) void svdq_wmma_kernel(
    const _Float16* __restrict__ x,
    const int*      __restrict__ qw,
    const _Float16* __restrict__ sc,
    const _Float16* __restrict__ lu,
    const _Float16* __restrict__ bias,
    const _Float16* __restrict__ t,
    float*          __restrict__ out)
{
  const int lane = threadIdx.x & 31;
  const int wave = threadIdx.x >> 5;
  const int tile = blockIdx.x * 8 + wave;   // 0..1791
  const int n0   = tile * 16;
  const int g    = lane >> 4;               // K-half select (0/1)
  const int col  = lane & 15;               // output column within tile
  const int N    = n0 + col;                // output feature (B/C/D column)
  const int m    = col;                     // token row (A row)

  v8f acc;
  // --- LoRA seed: acc = (x @ lora_down^T) @ lora_up^T tile ---
  {
    const _Float16* tp = t + m * RANK;
    v8h a0 = *(const v8h*)(tp + 16 * g);
    v8h a1 = *(const v8h*)(tp + 16 * g + 8);
    const _Float16* lp = lu + (size_t)N * RANK + 16 * g;
    v8h b0 = *(const v8h*)(lp);
    v8h b1 = *(const v8h*)(lp + 8);
    v16h A, B;
#pragma unroll
    for (int j = 0; j < 8; ++j) { A[j] = a0[j]; A[8+j] = a1[j];
                                  B[j] = b0[j]; B[8+j] = b1[j]; }
    v8f z = {};
    acc = __builtin_amdgcn_wmma_f32_16x16x32_f16(
        false, A, false, B, (short)0, z, false, false);
  }

  const _Float16* xr = x  + (size_t)m * IN_F;
  const int*      qr = qw + (size_t)N * IN_F + g * 16;
  const _Float16* sr = sc + (size_t)N * SCALES_PER_ROW;

  const v2h kM1032 = { (_Float16)(-1032), (_Float16)(-1032) };

#pragma unroll 2
  for (int k = 0; k < IN_F; k += GROUP) {
    _Float16 sh = sr[k >> 6];                // one scale per 64 K = 2 WMMAs
    v2h sv = { sh, sh };
#pragma unroll
    for (int h = 0; h < 2; ++h) {
      const int kk = k + 32 * h;
      // A tile of x (hot in L2/WGP$)
      v8h a0 = *(const v8h*)(xr + kk + 16 * g);
      v8h a1 = *(const v8h*)(xr + kk + 16 * g + 8);
      // B tile of qweight: 16 consecutive int32 per lane, nontemporal stream
      const v4i* qp = (const v4i*)(qr + kk);
      v4i qq0 = __builtin_nontemporal_load(qp + 0);
      v4i qq1 = __builtin_nontemporal_load(qp + 1);
      v4i qq2 = __builtin_nontemporal_load(qp + 2);
      v4i qq3 = __builtin_nontemporal_load(qp + 3);
      v4i qq[4] = { qq0, qq1, qq2, qq3 };

      v16h A;
#pragma unroll
      for (int j = 0; j < 8; ++j) { A[j] = a0[j]; A[8+j] = a1[j]; }

      v8u Bu;
#pragma unroll
      for (int p = 0; p < 8; ++p) {
        u32 qe = (u32)qq[p >> 1][(p & 1) * 2];
        u32 qo = (u32)qq[p >> 1][(p & 1) * 2 + 1];
        // halves = {byte0(qe), byte0(qo)}
        u32 pk = __builtin_amdgcn_perm(qo, qe, 0x0C040C00u);
        pk = (pk & 0x000F000Fu) ^ 0x64086408u;   // f16 pair = 1032 + q (exact)
        v2h hh = __builtin_bit_cast(v2h, pk);
        v2h w  = (hh + kM1032) * sv;             // exact sub, single f16 round
        Bu[p]  = __builtin_bit_cast(u32, w);
      }
      v16h B = __builtin_bit_cast(v16h, Bu);

      acc = __builtin_amdgcn_wmma_f32_16x16x32_f16(
          false, A, false, B, (short)0, acc, false, false);
    }
  }

  // C/D layout: element v -> row M = v + 8g, column N
  float bv = (float)bias[N];
#pragma unroll
  for (int v = 0; v < 8; ++v) {
    out[(size_t)(v + 8 * g) * OUT_F + N] = acc[v] + bv;
  }
}

// ---------------------------------------------------------------------------
// Inputs (setup_inputs order): x, qweight, wscales, lora_down, lora_up, bias
// ---------------------------------------------------------------------------
extern "C" void kernel_launch(void* const* d_in, const int* in_sizes, int n_in,
                              void* d_out, int out_size, void* d_ws, size_t ws_size,
                              hipStream_t stream) {
  const _Float16* x    = (const _Float16*)d_in[0];
  const int*      qw   = (const int*)d_in[1];
  const _Float16* wsc  = (const _Float16*)d_in[2];
  const _Float16* ld   = (const _Float16*)d_in[3];
  const _Float16* lu   = (const _Float16*)d_in[4];
  const _Float16* bias = (const _Float16*)d_in[5];
  float*     out = (float*)d_out;
  _Float16*  t   = (_Float16*)d_ws;            // 16x32 fp16 = 1 KB scratch

  lora_down_kernel<<<1, 512, 0, stream>>>(x, ld, t);

  const int tiles  = OUT_F / 16;               // 1792
  const int blocks = tiles / 8;                // 224 blocks x 8 waves
  svdq_wmma_kernel<<<blocks, 256, 0, stream>>>(x, qw, wsc, lu, bias, t, out);
}
